// GGNNSum_60275571032229
// MI455X (gfx1250) — matrix operand order repaired and checked by reading, not compiled, not run on hardware
//
#include <hip/hip_runtime.h>
#include <hip/hip_bf16.h>
#include <math.h>

// ---------------------------------------------------------------------------
// GGNN (GatedGraphConv + sum pool) for gfx1250.
// All GEMMs run on v_wmma_f32_16x16x32_bf16 (wave32, 16x16 tiles, K in steps
// of 32). Weights pre-converted to bf16 and laid out as B^T (row-major
// [out][in]) so every B fragment is two contiguous 16B loads per lane.
// k_transform processes M=32 per wave so each B fragment feeds two WMMAs.
// ---------------------------------------------------------------------------

typedef __attribute__((ext_vector_type(16))) __bf16 bf16x16;
typedef __attribute__((ext_vector_type(8)))  float  f32x8;

union FragAB { bf16x16 v; uint4 q[2]; unsigned short u[16]; };

#define DIM     128
#define KSTEPS  4     // 128 / 32
#define NTILES  8     // 128 / 16
#define NSTEPS  8

__device__ __forceinline__ unsigned short f2bf(float f) {
  unsigned int x = __float_as_uint(f);
  x += 0x7FFFu + ((x >> 16) & 1u);          // round-to-nearest-even
  return (unsigned short)(x >> 16);
}
__device__ __forceinline__ float bf2f(unsigned short u) {
  return __uint_as_float(((unsigned int)u) << 16);
}

__device__ __forceinline__ f32x8 wmma_bf16(bf16x16 a, bf16x16 b, f32x8 c) {
  // D = A(16x32) * B(32x16) + C(16x16), f32 accumulate
  return __builtin_amdgcn_wmma_f32_16x16x32_bf16(
      /*neg_a=*/false, a, /*neg_b=*/false, b,
      /*c_mod=*/(short)0, c, /*reuse_a=*/false, /*reuse_b=*/false);
}

// A fragment (16x32 bf16 tile) from row-major bf16 matrix, tile origin (m0,k0).
// ISA layout: lane m = lane&15; VGPR0-3 hold K = half*8 .. +7,
// VGPR4-7 hold K = 16+half*8 .. +7  -> two contiguous 16B loads.
__device__ __forceinline__ bf16x16 ldfrag_a16(const unsigned short* __restrict__ A,
                                              int lda, int m0, int k0, int lane) {
  int m = m0 + (lane & 15), half = lane >> 4;
  const unsigned short* p = A + (size_t)m * lda + k0 + half * 8;
  FragAB f;
  f.q[0] = *(const uint4*)p;          // K = k0 + half*8 + [0..7]
  f.q[1] = *(const uint4*)(p + 16);   // K = k0 + 16 + half*8 + [0..7]
  return f.v;
}

// A fragment from an f32 source, converting to bf16 in-register.
__device__ __forceinline__ bf16x16 ldfrag_a32(const float* __restrict__ A,
                                              int lda, int m0, int k0, int lane) {
  int m = m0 + (lane & 15), half = lane >> 4;
  const float* row = A + (size_t)m * lda + k0 + half * 8;
  FragAB f;
#pragma unroll
  for (int g = 0; g < 2; ++g) {
    float4 x0 = *(const float4*)(row + g * 16);
    float4 x1 = *(const float4*)(row + g * 16 + 4);
    f.u[g*8+0] = f2bf(x0.x); f.u[g*8+1] = f2bf(x0.y);
    f.u[g*8+2] = f2bf(x0.z); f.u[g*8+3] = f2bf(x0.w);
    f.u[g*8+4] = f2bf(x1.x); f.u[g*8+5] = f2bf(x1.y);
    f.u[g*8+6] = f2bf(x1.z); f.u[g*8+7] = f2bf(x1.w);
  }
  return f.v;
}

// B fragment (32x16) from B^T stored row-major [outRows][ldk]; tile (k0, n0).
// Lane n = lane&15 reads row n at K = k0+half*8+[0..7] and k0+16+half*8+[0..7].
__device__ __forceinline__ bf16x16 ldfrag_bt(const unsigned short* __restrict__ BT,
                                             int ldk, int n0, int k0, int lane) {
  int n = n0 + (lane & 15), half = lane >> 4;
  const unsigned short* p = BT + (size_t)n * ldk + k0 + half * 8;
  FragAB f;
  f.q[0] = *(const uint4*)p;
  f.q[1] = *(const uint4*)(p + 16);
  return f.v;
}

// ---------------------------------------------------------------------------
// ht16[e][n][:] = bf16( h16[n][:] @ W_edge[e] )
// 32 rows per wave: each B fragment is consumed by two WMMAs back-to-back.
// ---------------------------------------------------------------------------
__global__ void k_transform(const unsigned short* __restrict__ h16,
                            const unsigned short* __restrict__ WeT,   // [E][out][in] bf16
                            unsigned short* __restrict__ ht16, int N) {
  int lane = threadIdx.x & 31;
  int wave = threadIdx.x >> 5;
  int row0 = (blockIdx.x * (blockDim.x >> 5) + wave) * 32;
  if (row0 >= N) return;                       // wave-uniform tail guard (N % 32 == 0)
  int e = blockIdx.y;
  const unsigned short* WT = WeT + (size_t)e * DIM * DIM;

  bf16x16 af[2][KSTEPS];                       // two 16-row A tile sets
#pragma unroll
  for (int mi = 0; mi < 2; ++mi)
#pragma unroll
    for (int kk = 0; kk < KSTEPS; ++kk)
      af[mi][kk] = ldfrag_a16(h16, DIM, row0 + mi * 16, kk * 32, lane);

  int half = lane >> 4, nlo = lane & 15;
  unsigned short* obase = ht16 + (size_t)e * N * DIM;
#pragma unroll
  for (int t = 0; t < NTILES; ++t) {
    f32x8 acc0 = {0.f,0.f,0.f,0.f,0.f,0.f,0.f,0.f};
    f32x8 acc1 = {0.f,0.f,0.f,0.f,0.f,0.f,0.f,0.f};
#pragma unroll
    for (int kk = 0; kk < KSTEPS; ++kk) {
      bf16x16 b = ldfrag_bt(WT, DIM, t * 16, kk * 32, lane);
      acc0 = wmma_bf16(af[0][kk], b, acc0);    // same B, two M tiles
      acc1 = wmma_bf16(af[1][kk], b, acc1);
    }
#pragma unroll
    for (int r = 0; r < 8; ++r) {
      int row = row0 + r + 8 * half;           // C layout: M = vgpr + 8*half, N = lane&15
      obase[(size_t)row * DIM + t * 16 + nlo]        = f2bf(acc0[r]);
      obase[(size_t)(row + 16) * DIM + t * 16 + nlo] = f2bf(acc1[r]);
    }
  }
}

// ---------------------------------------------------------------------------
// Edge scatter: a[dst] += ht16[etype][src]  (one wave per edge, 4 cols/lane)
// ---------------------------------------------------------------------------
__global__ void k_edges(const unsigned short* __restrict__ ht16,
                        const int* __restrict__ esrc, const int* __restrict__ edst,
                        const int* __restrict__ etyp,
                        float* __restrict__ aggr, int E, int N) {
  long long t = (long long)blockIdx.x * blockDim.x + threadIdx.x;
  int edge = (int)(t >> 5);
  if (edge >= E) return;
  int lane = (int)(t & 31);
  int src = esrc[edge], dst = edst[edge], e = etyp[edge];
  const unsigned short* m = ht16 + ((size_t)e * N + src) * DIM + lane * 4;
  uint2 raw = *(const uint2*)m;                     // 8B/lane, 256B/wave coalesced
  float* arow = aggr + (size_t)dst * DIM + lane * 4;
  atomicAdd(arow + 0, bf2f((unsigned short)(raw.x & 0xFFFFu)));
  atomicAdd(arow + 1, bf2f((unsigned short)(raw.x >> 16)));
  atomicAdd(arow + 2, bf2f((unsigned short)(raw.y & 0xFFFFu)));
  atomicAdd(arow + 3, bf2f((unsigned short)(raw.y >> 16)));
}

// ---------------------------------------------------------------------------
// Fused GRU: gi = a@Wih^T, gh = h@Whh^T, gates + blend, h updated in place.
// r/z accumulators take BOTH products (gi+gh fused in the WMMA C operand).
// ---------------------------------------------------------------------------
__global__ void k_gru(const float* __restrict__ aggr,
                      float* __restrict__ h,
                      const unsigned short* __restrict__ h16,
                      const unsigned short* __restrict__ Wih,   // [384][128] bf16 (== B^T)
                      const unsigned short* __restrict__ Whh,
                      const float* __restrict__ b_ih,
                      const float* __restrict__ b_hh, int N) {
  int lane = threadIdx.x & 31;
  int wave = threadIdx.x >> 5;
  int row0 = (blockIdx.x * (blockDim.x >> 5) + wave) * 16;
  if (row0 >= N) return;

  bf16x16 Aa[KSTEPS], Ah[KSTEPS];
#pragma unroll
  for (int kk = 0; kk < KSTEPS; ++kk) {
    Aa[kk] = ldfrag_a32(aggr, DIM, row0, kk * 32, lane);
    Ah[kk] = ldfrag_a16(h16,  DIM, row0, kk * 32, lane);
  }

  int half = lane >> 4, nlo = lane & 15;
#pragma unroll
  for (int t = 0; t < NTILES; ++t) {
    int col = t * 16 + nlo;
    f32x8 ar  = {0.f,0.f,0.f,0.f,0.f,0.f,0.f,0.f};
    f32x8 az  = {0.f,0.f,0.f,0.f,0.f,0.f,0.f,0.f};
    f32x8 ain = {0.f,0.f,0.f,0.f,0.f,0.f,0.f,0.f};
    f32x8 ahn = {0.f,0.f,0.f,0.f,0.f,0.f,0.f,0.f};
#pragma unroll
    for (int kk = 0; kk < KSTEPS; ++kk) {
      int k0 = kk * 32;
      ar  = wmma_bf16(Aa[kk], ldfrag_bt(Wih, DIM,           t*16, k0, lane), ar);
      ar  = wmma_bf16(Ah[kk], ldfrag_bt(Whh, DIM,           t*16, k0, lane), ar);
      az  = wmma_bf16(Aa[kk], ldfrag_bt(Wih, DIM,   DIM   + t*16, k0, lane), az);
      az  = wmma_bf16(Ah[kk], ldfrag_bt(Whh, DIM,   DIM   + t*16, k0, lane), az);
      ain = wmma_bf16(Aa[kk], ldfrag_bt(Wih, DIM, 2*DIM   + t*16, k0, lane), ain);
      ahn = wmma_bf16(Ah[kk], ldfrag_bt(Whh, DIM, 2*DIM   + t*16, k0, lane), ahn);
    }
    float br = b_ih[col]           + b_hh[col];
    float bz = b_ih[DIM + col]     + b_hh[DIM + col];
    float bi = b_ih[2 * DIM + col];
    float bh = b_hh[2 * DIM + col];
#pragma unroll
    for (int r = 0; r < 8; ++r) {
      size_t idx = (size_t)(row0 + r + 8 * half) * DIM + col;
      float rr = 1.f / (1.f + __expf(-(ar[r] + br)));
      float zz = 1.f / (1.f + __expf(-(az[r] + bz)));
      float nn = tanhf(ain[r] + bi + rr * (ahn[r] + bh));
      float hv = h[idx];
      h[idx] = (1.f - zz) * nn + zz * hv;        // row-local: in-place safe
    }
  }
}

// ---------------------------------------------------------------------------
// Small helper kernels
// ---------------------------------------------------------------------------
__global__ void k_f32_to_bf16(const float* __restrict__ s,
                              unsigned short* __restrict__ d, long long n) {
  long long i = ((long long)blockIdx.x * blockDim.x + threadIdx.x) * 4;
  if (i + 3 < n) {
    float4 x = *(const float4*)(s + i);
    ushort4 o;
    o.x = f2bf(x.x); o.y = f2bf(x.y); o.z = f2bf(x.z); o.w = f2bf(x.w);
    *(ushort4*)(d + i) = o;
  } else {
    for (; i < n; ++i) d[i] = f2bf(s[i]);
  }
}

// WeT[e][o][d] = bf16( W_edge[e][d][o] )  — transpose to B^T layout
__global__ void k_prep_wedge(const float* __restrict__ We,
                             unsigned short* __restrict__ WeT, int ne) {
  int i = blockIdx.x * blockDim.x + threadIdx.x;
  if (i >= ne * DIM * DIM) return;
  int e = i / (DIM * DIM);
  int rem = i - e * DIM * DIM;
  int o = rem / DIM, d = rem - o * DIM;
  WeT[i] = f2bf(We[(size_t)e * DIM * DIM + (size_t)d * DIM + o]);
}

__global__ void k_zero(float* __restrict__ p, long long n) {
  long long i = ((long long)blockIdx.x * blockDim.x + threadIdx.x) * 4;
  if (i + 3 < n) {
    *(float4*)(p + i) = make_float4(0.f, 0.f, 0.f, 0.f);
  } else {
    for (; i < n; ++i) p[i] = 0.f;
  }
}

__global__ void k_pool(const float* __restrict__ h, const int* __restrict__ gid,
                       float* __restrict__ pooled, int N) {
  long long i = (long long)blockIdx.x * blockDim.x + threadIdx.x;
  if (i >= (long long)N * DIM) return;
  int node = (int)(i >> 7), c = (int)(i & (DIM - 1));
  atomicAdd(&pooled[gid[node] * DIM + c], h[i]);   // pooled (64KB) is L2-resident
}

__global__ void k_cls(const float* __restrict__ pooled, const float* __restrict__ Wc,
                      const float* __restrict__ bc, float* __restrict__ out, int G) {
  int g = threadIdx.x;
  if (g >= G) return;
  float s = bc[0];
#pragma unroll 4
  for (int d = 0; d < DIM; ++d) s += pooled[g * DIM + d] * Wc[d];
  out[g] = 1.f / (1.f + __expf(-s));
}

// ---------------------------------------------------------------------------
// Host launcher
// ---------------------------------------------------------------------------
extern "C" void kernel_launch(void* const* d_in, const int* in_sizes, int n_in,
                              void* d_out, int out_size, void* d_ws, size_t ws_size,
                              hipStream_t stream) {
  const float* features = (const float*)d_in[0];
  const int*   esrc     = (const int*)d_in[1];
  const int*   edst     = (const int*)d_in[2];
  const int*   etyp     = (const int*)d_in[3];
  const int*   gid      = (const int*)d_in[4];
  const float* W_edge   = (const float*)d_in[5];
  const float* W_ih     = (const float*)d_in[6];
  const float* W_hh     = (const float*)d_in[7];
  const float* b_ih     = (const float*)d_in[8];
  const float* b_hh     = (const float*)d_in[9];
  const float* W_cls    = (const float*)d_in[10];
  const float* b_cls    = (const float*)d_in[11];
  float* out = (float*)d_out;

  const int N  = in_sizes[4];                  // 100000 (divisible by 32)
  const int E  = in_sizes[1];                  // 600000
  const int G  = out_size;                     // 128 graphs
  const int NE = in_sizes[5] / (DIM * DIM);    // 4 edge types

  // ---- workspace carve-up (256B aligned) ----
  char* ws = (char*)d_ws;
  size_t off = 0;
  auto carve = [&](size_t bytes) -> char* {
    char* p = ws + off;
    off += (bytes + 255) & ~(size_t)255;
    return p;
  };
  float*          h      = (float*)         carve((size_t)N * DIM * 4);
  unsigned short* h16    = (unsigned short*)carve((size_t)N * DIM * 2);
  unsigned short* ht16   = (unsigned short*)carve((size_t)NE * N * DIM * 2);
  float*          aggr   = (float*)         carve((size_t)N * DIM * 4);
  unsigned short* WeT16  = (unsigned short*)carve((size_t)NE * DIM * DIM * 2);
  unsigned short* Wih16  = (unsigned short*)carve((size_t)3 * DIM * DIM * 2);
  unsigned short* Whh16  = (unsigned short*)carve((size_t)3 * DIM * DIM * 2);
  float*          pooled = (float*)         carve((size_t)G * DIM * 4);
  (void)ws_size; (void)n_in;

  const int THREADS = 256;
  auto g4 = [&](long long n) { return (int)((n / 4 + THREADS - 1) / THREADS); };

  // ---- one-time weight prep (re-done each launch: deterministic) ----
  k_f32_to_bf16<<<g4(3LL * DIM * DIM), THREADS, 0, stream>>>(W_ih, Wih16, 3LL * DIM * DIM);
  k_f32_to_bf16<<<g4(3LL * DIM * DIM), THREADS, 0, stream>>>(W_hh, Whh16, 3LL * DIM * DIM);
  k_prep_wedge<<<(NE * DIM * DIM + THREADS - 1) / THREADS, THREADS, 0, stream>>>(W_edge, WeT16, NE);

  // h0 = features (IN_DIM == OUT_DIM, no pad); never mutate the input buffer
  hipMemcpyAsync(h, features, (size_t)N * DIM * sizeof(float),
                 hipMemcpyDeviceToDevice, stream);

  const long long ND = (long long)N * DIM;
  dim3 tblk(128);                                    // 4 waves x 32 rows = 128 rows/block
  dim3 tgrid((N + 127) / 128, NE);
  dim3 ggrid((N + 63) / 64);                         // 4 waves x 16 rows = 64 rows/block
  int  egrid = (int)(((long long)E * 32 + THREADS - 1) / THREADS);

  for (int step = 0; step < NSTEPS; ++step) {
    k_f32_to_bf16<<<g4(ND), THREADS, 0, stream>>>(h, h16, ND);
    k_transform<<<tgrid, tblk, 0, stream>>>(h16, WeT16, ht16, N);
    k_zero<<<g4(ND), THREADS, 0, stream>>>(aggr, ND);
    k_edges<<<egrid, THREADS, 0, stream>>>(ht16, esrc, edst, etyp, aggr, E, N);
    k_gru<<<ggrid, tblk, 0, stream>>>(aggr, h, h16, Wih16, Whh16, b_ih, b_hh, N);
  }

  k_zero<<<g4((long long)G * DIM), THREADS, 0, stream>>>(pooled, (long long)G * DIM);
  k_pool<<<(int)((ND + THREADS - 1) / THREADS), THREADS, 0, stream>>>(h, gid, pooled, N);
  k_cls<<<1, THREADS, 0, stream>>>(pooled, W_cls, b_cls, out, G);
}
